// Router_58969900974703
// MI455X (gfx1250) — matrix operand order repaired and checked by reading, not compiled, not run on hardware
//
#include <hip/hip_runtime.h>
#include <math.h>

// ---------------------------------------------------------------------------
// MoE router for MI455X (gfx1250, wave32).
//   logits = x[16384,4096] @ gate_w[64,4096]^T via v_wmma_f32_16x16x32_bf16,
//   gate_w staged bf16 in LDS with double-buffered global_load_async_to_lds,
//   then per-token top-2 + softmax + expert histogram + load-balance loss.
//
// ws layout: [0 .. 512KB)  bf16 copy of gate_w ; then float expert_counts[64]
// d_out:     [0,32768) routing_weights, [32768,65536) indices(f32), [65536] loss
// ---------------------------------------------------------------------------

#define TOKENS  16384
#define DMODEL  4096
#define NEXP    64
#define WAVES   4            // 128 threads/block, 4 wave32
#define KC      128          // K-chunk staged in LDS per buffer
#define BSTRIDE 136          // LDS row stride (halfs) = 68 dwords == 4 mod 64
#define NCHUNK  (DMODEL / KC)

typedef __attribute__((ext_vector_type(16))) __bf16 v16bf;
typedef __attribute__((ext_vector_type(8)))  float  v8f;

// --- gfx1250 async copy: global -> LDS, 16 bytes, tracked by ASYNCcnt -------
__device__ __forceinline__ void async_load_b128(unsigned lds_off, const void* gptr) {
    asm volatile("global_load_async_to_lds_b128 %0, %1, off"
                 :: "v"(lds_off), "v"(gptr) : "memory");
}
__device__ __forceinline__ void wait_async0() {
    asm volatile("s_wait_asynccnt 0x0" ::: "memory");
}

// --- kernel 1: gate_w f32 -> bf16 into ws; zero the histogram ---------------
__global__ void cvt_gate_kernel(const float* __restrict__ gw,
                                __bf16* __restrict__ gwb,
                                float* __restrict__ counts) {
    int i = blockIdx.x * blockDim.x + threadIdx.x;
    if (i < NEXP * DMODEL) gwb[i] = (__bf16)gw[i];
    if (blockIdx.x == 0 && threadIdx.x < NEXP) counts[threadIdx.x] = 0.0f;
}

// --- kernel 2: fused GEMM (bf16 WMMA) + top-2 + softmax + histogram ---------
__global__ __launch_bounds__(WAVES * 32)
void router_gemm_kernel(const float* __restrict__ x,
                        const __bf16* __restrict__ gwb,
                        float* __restrict__ out,
                        float* __restrict__ counts) {
    // double-buffered B staging: 2 x 64 experts x KC halfs (padded) = 2x17KB
    __shared__ __align__(16) __bf16 smemB[2][NEXP * BSTRIDE];

    const int tid  = threadIdx.x;
    const int lane = tid & 31;
    const int wave = tid >> 5;
    const int nrow = lane & 15;          // token row (A/C) / expert col (B)
    const int kg   = lane >> 4;          // K half-group
    const int mtile = blockIdx.x * WAVES + wave;

    const float* xrow = x + (size_t)(mtile * 16 + nrow) * DMODEL;

    // stage one K-chunk of bf16 gate_w into LDS buffer `buf` (async, 8x16B/thread)
    auto stage = [&](int buf, int kb) {
        const __bf16* src = gwb + kb;
        unsigned lbase = (unsigned)(size_t)&smemB[buf][0];
#pragma unroll
        for (int i = 0; i < 8; ++i) {
            int u  = tid + i * 128;          // 0..1023 16B-units
            int e  = u >> 4;                 // expert row 0..63
            int c8 = (u & 15) * 8;           // half offset in chunk 0..120
            async_load_b128(lbase + (unsigned)(e * BSTRIDE + c8) * 2u,
                            src + (size_t)e * DMODEL + c8);
        }
    };

    v8f acc0 = {}, acc1 = {}, acc2 = {}, acc3 = {};

    stage(0, 0);
    wait_async0();
    __syncthreads();

    for (int ch = 0; ch < NCHUNK; ++ch) {
        if (ch + 1 < NCHUNK) stage((ch + 1) & 1, (ch + 1) * KC);  // overlap w/ compute

        const __bf16* bbase = &smemB[ch & 1][0];
        const int kb = ch * KC;
#pragma unroll
        for (int k0 = 0; k0 < KC; k0 += 32) {
            // A tile 16x32 bf16: lane holds 16 consecutive K
            const float* ap = xrow + kb + k0 + kg * 16;
            float4 a0 = *(const float4*)(ap);
            float4 a1 = *(const float4*)(ap + 4);
            float4 a2 = *(const float4*)(ap + 8);
            float4 a3 = *(const float4*)(ap + 12);
            v16bf a;
            a[0]  = (__bf16)a0.x; a[1]  = (__bf16)a0.y; a[2]  = (__bf16)a0.z; a[3]  = (__bf16)a0.w;
            a[4]  = (__bf16)a1.x; a[5]  = (__bf16)a1.y; a[6]  = (__bf16)a1.z; a[7]  = (__bf16)a1.w;
            a[8]  = (__bf16)a2.x; a[9]  = (__bf16)a2.y; a[10] = (__bf16)a2.z; a[11] = (__bf16)a2.w;
            a[12] = (__bf16)a3.x; a[13] = (__bf16)a3.y; a[14] = (__bf16)a3.z; a[15] = (__bf16)a3.w;

            // B tiles 32x16 bf16 from LDS: lane = expert col, 16 consecutive K
            const __bf16* bp = bbase + k0 + kg * 16;
            v16bf b0 = *(const v16bf*)(bp + (0 * 16 + nrow) * BSTRIDE);
            v16bf b1 = *(const v16bf*)(bp + (1 * 16 + nrow) * BSTRIDE);
            v16bf b2 = *(const v16bf*)(bp + (2 * 16 + nrow) * BSTRIDE);
            v16bf b3 = *(const v16bf*)(bp + (3 * 16 + nrow) * BSTRIDE);

            acc0 = __builtin_amdgcn_wmma_f32_16x16x32_bf16(false, a, false, b0,
                                                           (short)0, acc0, false, false);
            acc1 = __builtin_amdgcn_wmma_f32_16x16x32_bf16(false, a, false, b1,
                                                           (short)0, acc1, false, false);
            acc2 = __builtin_amdgcn_wmma_f32_16x16x32_bf16(false, a, false, b2,
                                                           (short)0, acc2, false, false);
            acc3 = __builtin_amdgcn_wmma_f32_16x16x32_bf16(false, a, false, b3,
                                                           (short)0, acc3, false, false);
        }
        wait_async0();        // next chunk staged
        __syncthreads();      // everyone done reading current chunk
    }

    // ---- epilogue: reuse LDS buffer 0 for logits scratch -------------------
    float* sw = (float*)&smemB[0][0] + wave * 16 * 65;
#pragma unroll
    for (int j = 0; j < 8; ++j) {
        int m = j + 8 * kg;               // C layout: VGPR j -> M = j + 8*kg, N = nrow
        sw[m * 65 + nrow +  0] = acc0[j];
        sw[m * 65 + nrow + 16] = acc1[j];
        sw[m * 65 + nrow + 32] = acc2[j];
        sw[m * 65 + nrow + 48] = acc3[j];
    }
    // same-wave producer/consumer; compiler inserts s_wait_dscnt

    if (lane < 16) {
        const float* r = sw + lane * 65;
        float best = -3.402823466e38f, sec = -3.402823466e38f;
        int bi = 0, si = 0;
        for (int n = 0; n < NEXP; ++n) {
            float v = r[n];
            if (v > best)     { sec = best; si = bi; best = v; bi = n; }
            else if (v > sec) { sec = v; si = n; }
        }
        float e   = __expf(sec - best);
        float inv = 1.0f / (1.0f + e);
        int tok = mtile * 16 + lane;
        out[tok * 2 + 0] = inv;
        out[tok * 2 + 1] = e * inv;
        out[2 * TOKENS + tok * 2 + 0] = (float)bi;
        out[2 * TOKENS + tok * 2 + 1] = (float)si;
        atomicAdd(&counts[bi], 1.0f);
        atomicAdd(&counts[si], 1.0f);
    }
}

// --- kernel 3: loss = 0.01 * std(counts, ddof=1)/(mean+1e-6) ----------------
__global__ void loss_kernel(const float* __restrict__ counts,
                            float* __restrict__ out) {
    if (blockIdx.x == 0 && threadIdx.x == 0) {
        float s = 0.0f;
        for (int i = 0; i < NEXP; ++i) s += counts[i];
        float mean = s / (float)NEXP;
        float ss = 0.0f;
        for (int i = 0; i < NEXP; ++i) {
            float d = counts[i] - mean;
            ss += d * d;
        }
        float stdv = sqrtf(ss / (float)(NEXP - 1));
        out[4 * TOKENS] = stdv / (mean + 1e-6f) * 0.01f;
    }
}

extern "C" void kernel_launch(void* const* d_in, const int* in_sizes, int n_in,
                              void* d_out, int out_size, void* d_ws, size_t ws_size,
                              hipStream_t stream) {
    const float* x  = (const float*)d_in[0];   // [16384, 4096] f32
    const float* gw = (const float*)d_in[1];   // [64, 4096] f32
    float* out = (float*)d_out;

    __bf16* gwb   = (__bf16*)d_ws;
    float* counts = (float*)((char*)d_ws + (size_t)NEXP * DMODEL * sizeof(__bf16));

    cvt_gate_kernel<<<(NEXP * DMODEL + 255) / 256, 256, 0, stream>>>(gw, gwb, counts);
    router_gemm_kernel<<<TOKENS / 16 / WAVES, WAVES * 32, 0, stream>>>(x, gwb, out, counts);
    loss_kernel<<<1, 64, 0, stream>>>(counts, out);
}
